// MultiheadCosformerAttention_63299228009161
// MI455X (gfx1250) — matrix operand order repaired and checked by reading, not compiled, not run on hardware
//
#include <hip/hip_runtime.h>
#include <hip/hip_bf16.h>

// ---------------------------------------------------------------------------
// cosFormer attention, MI455X / gfx1250.
// All GEMMs on v_wmma_f32_16x16x32_bf16; fp32->bf16 conversion hoisted into
// one-time bandwidth-bound pre-pass kernels so GEMM loops are load+wmma only.
// L=2048, B=8, E=1024, H=16, D=64, 2D=128
// ---------------------------------------------------------------------------

typedef __attribute__((ext_vector_type(16))) __bf16 v16bf;
typedef __attribute__((ext_vector_type(8)))  float  v8f;

#define WMMA_BF16(a, b, c) \
    __builtin_amdgcn_wmma_f32_16x16x32_bf16(false, (a), false, (b), (short)0, (c), false, false)

#define CF_L   2048
#define CF_B   8
#define CF_E   1024
#define CF_H   16
#define CF_D   64
#define CF_2D  128
#define CF_3E  3072
#define CF_ROWS (CF_L * CF_B)      // 16384
#define CF_BH   (CF_B * CF_H)      // 128

__device__ __forceinline__ __bf16 f2bf(float f) {
    unsigned u = __float_as_uint(f);
    u += 0x7fffu + ((u >> 16) & 1u);            // round-to-nearest-even
    unsigned short s = (unsigned short)(u >> 16);
    return __builtin_bit_cast(__bf16, s);
}

__device__ __forceinline__ float bf2f(__bf16 x) {
    unsigned short s = __builtin_bit_cast(unsigned short, x);
    return __uint_as_float(((unsigned)s) << 16);
}

// A-operand (16x32 bf16, M x K): lane holds row M=lane&15; two contiguous
// 8-element K chunks at kb and kb+16, kb = (lane>=16)*8.  p points at
// &A[M*lda + kb].
__device__ __forceinline__ v16bf load_a_bf(const __bf16* p) {
    union { v16bf v; uint4 q[2]; } u;
    u.q[0] = *(const uint4*)(p);
    u.q[1] = *(const uint4*)(p + 16);
    return u.v;
}

// B-operand (32x16 bf16, K x N) loaded from B^T row-major: lane holds column
// N=lane&15 as two contiguous 8-element K chunks at kb and kb+16,
// kb=(lane>=16)*8.  p points at &BT[N*ldb + kb].
__device__ __forceinline__ v16bf load_b_bf(const __bf16* p) {
    union { v16bf v; uint4 q[2]; } u;
    u.q[0] = *(const uint4*)(p);
    u.q[1] = *(const uint4*)(p + 16);
    return u.v;
}

// ---------------------------------------------------------------------------
// 0) fp32 -> bf16 pre-conversion (vectorized, bandwidth bound, one pass)
// ---------------------------------------------------------------------------
__global__ void cvt_bf16_kernel(const float* __restrict__ src,
                                __bf16* __restrict__ dst, int n) {
    const int i = (blockIdx.x * 256 + threadIdx.x) * 8;
    if (i + 8 > n) return;
    float4 f0 = *(const float4*)(src + i);
    float4 f1 = *(const float4*)(src + i + 4);
    union { __bf16 h[8]; uint4 q; } u;
    u.h[0] = f2bf(f0.x); u.h[1] = f2bf(f0.y); u.h[2] = f2bf(f0.z); u.h[3] = f2bf(f0.w);
    u.h[4] = f2bf(f1.x); u.h[5] = f2bf(f1.y); u.h[6] = f2bf(f1.z); u.h[7] = f2bf(f1.w);
    *(uint4*)(dst + i) = u.q;
}

// ---------------------------------------------------------------------------
// 1) QKV projection: qkv = X(16384x1024) * W^T(1024x3072) + bias
//    One wave = 32x64 block (2 Mtiles x 4 Ntiles): 8 wmma per K-step.
//    Fused: bias + ReLU(q,k) + cosFormer sin/cos + scatter to stage-2 layouts.
//    512 row-strips x 48 col-strips = 24576 waves.
// ---------------------------------------------------------------------------
__global__ void qkv_kernel(const __bf16* __restrict__ X,
                           const __bf16* __restrict__ W,
                           const float* __restrict__ bias,
                           __bf16* __restrict__ qb,    // [bh][l][128]
                           __bf16* __restrict__ kt,    // [bh][128][L]
                           __bf16* __restrict__ vt) {  // [bh][64][L]
    const int lane = threadIdx.x & 31;
    const int wid  = blockIdx.x * 8 + (threadIdx.x >> 5);
    const int ms   = wid / 48;         // 0..511  (32-row strip)
    const int ns   = wid % 48;         // 0..47   (64-col strip)

    const __bf16* A0 = X + (size_t)(ms * 32 +      (lane & 15)) * CF_E + ((lane >> 4) * 8);
    const __bf16* A1 = X + (size_t)(ms * 32 + 16 + (lane & 15)) * CF_E + ((lane >> 4) * 8);
    const __bf16* Bt = W + (size_t)(ns * 64 + (lane & 15)) * CF_E + ((lane >> 4) * 8);

    v8f acc[2][4] = {{v8f{}, v8f{}, v8f{}, v8f{}}, {v8f{}, v8f{}, v8f{}, v8f{}}};
    for (int k0 = 0; k0 < CF_E; k0 += 32) {
        v16bf a0 = load_a_bf(A0 + k0);
        v16bf a1 = load_a_bf(A1 + k0);
#pragma unroll
        for (int nt = 0; nt < 4; ++nt) {
            v16bf b = load_b_bf(Bt + (size_t)nt * 16 * CF_E + k0);
            acc[0][nt] = WMMA_BF16(a0, b, acc[0][nt]);
            acc[1][nt] = WMMA_BF16(a1, b, acc[1][nt]);
        }
    }

#pragma unroll
    for (int nt = 0; nt < 4; ++nt) {
        const int col = ns * 64 + nt * 16 + (lane & 15);   // 0..3071
        const float bv = bias[col];
        const int sec = col >> 10;           // 0=q 1=k 2=v
        const int e   = col & 1023;
        const int h   = e >> 6;
        const int d   = e & 63;
#pragma unroll
        for (int mi = 0; mi < 2; ++mi) {
#pragma unroll
            for (int r = 0; r < 8; ++r) {
                const int row = ms * 32 + mi * 16 + ((lane < 16) ? r : 8 + r); // l*B + b
                const int l = row >> 3;
                const int bb = row & 7;
                const int bh = bb * CF_H + h;
                float val = acc[mi][nt][r] + bv;
                if (sec == 2) {
                    vt[((size_t)bh * CF_D + d) * CF_L + l] = f2bf(val);
                } else {
                    val = fmaxf(val, 0.0f);
                    const float th = 1.5707963267948966f * (float)(l + 1) * (1.0f / (float)CF_L);
                    const float sn = __sinf(th), cs = __cosf(th);
                    if (sec == 0) {
                        __bf16* qrow = qb + ((size_t)bh * CF_L + l) * CF_2D;
                        qrow[d]        = f2bf(val * sn);
                        qrow[CF_D + d] = f2bf(val * cs);
                    } else {
                        __bf16* kc = kt + (size_t)bh * CF_2D * CF_L;
                        kc[(size_t)d * CF_L + l]          = f2bf(val * sn);
                        kc[(size_t)(CF_D + d) * CF_L + l] = f2bf(val * cs);
                    }
                }
            }
        }
    }
}

// ---------------------------------------------------------------------------
// 2) ksum[bh][m] = sum_l kt[bh][m][l]   (rows contiguous, wave per row)
// ---------------------------------------------------------------------------
__global__ void ksum_kernel(const __bf16* __restrict__ kt, float* __restrict__ ksum) {
    const int lane = threadIdx.x & 31;
    const int row  = blockIdx.x * 8 + (threadIdx.x >> 5);   // 0..16383
    const __bf16* p = kt + (size_t)row * CF_L;
    float s = 0.0f;
    for (int k = lane; k < CF_L; k += 32) s += bf2f(p[k]);
#pragma unroll
    for (int off = 16; off; off >>= 1) s += __shfl_xor(s, off, 32);
    if (lane == 0) ksum[row] = s;
}

// ---------------------------------------------------------------------------
// 3) kv[bh] = kt(128x2048) * vt^T(2048x64), output transposed kvt[bh][d][m].
//    One block per head, 8 waves each computing a 32x32 block (2x2 tiles):
//    4 loads : 4 wmma per K-step, K=2048.
// ---------------------------------------------------------------------------
__global__ void kv_kernel(const __bf16* __restrict__ kt,
                          const __bf16* __restrict__ vt,
                          __bf16* __restrict__ kvt) {
    const int lane = threadIdx.x & 31;
    const int w    = threadIdx.x >> 5;
    const int mp   = w >> 1;      // 0..3  (pair of Mtiles, m = 2D)
    const int np   = w & 1;       // 0..1  (pair of Ntiles, d = D)
    const int bh   = blockIdx.x;

    const __bf16* A0 = kt + (size_t)bh * CF_2D * CF_L
                          + (size_t)(mp * 32 +      (lane & 15)) * CF_L + ((lane >> 4) * 8);
    const __bf16* A1 = A0 + (size_t)16 * CF_L;
    const __bf16* B0 = vt + (size_t)bh * CF_D * CF_L
                          + (size_t)(np * 32 +      (lane & 15)) * CF_L + ((lane >> 4) * 8);
    const __bf16* B1 = B0 + (size_t)16 * CF_L;

    v8f acc[2][2] = {{v8f{}, v8f{}}, {v8f{}, v8f{}}};
    for (int k0 = 0; k0 < CF_L; k0 += 32) {
        v16bf a0 = load_a_bf(A0 + k0);
        v16bf a1 = load_a_bf(A1 + k0);
        v16bf b0 = load_b_bf(B0 + k0);
        v16bf b1 = load_b_bf(B1 + k0);
        acc[0][0] = WMMA_BF16(a0, b0, acc[0][0]);
        acc[0][1] = WMMA_BF16(a0, b1, acc[0][1]);
        acc[1][0] = WMMA_BF16(a1, b0, acc[1][0]);
        acc[1][1] = WMMA_BF16(a1, b1, acc[1][1]);
    }

    __bf16* out = kvt + (size_t)bh * CF_D * CF_2D;
#pragma unroll
    for (int ni = 0; ni < 2; ++ni) {
        const int d = np * 32 + ni * 16 + (lane & 15);
#pragma unroll
        for (int mi = 0; mi < 2; ++mi) {
#pragma unroll
            for (int r = 0; r < 8; ++r) {
                const int m = mp * 32 + mi * 16 + ((lane < 16) ? r : 8 + r);
                out[(size_t)d * CF_2D + m] = f2bf(acc[mi][ni][r]);
            }
        }
    }
}

// ---------------------------------------------------------------------------
// 4) z[bh][l] = 1 / max(q_[bh,l,:] . ksum[bh,:], eps)    (wave per row)
// ---------------------------------------------------------------------------
__global__ void z_kernel(const __bf16* __restrict__ qb,
                         const float* __restrict__ ksum,
                         float* __restrict__ z) {
    const int lane = threadIdx.x & 31;
    const int row  = blockIdx.x * 8 + (threadIdx.x >> 5);   // 0..262143 = bh*L+l
    const int bh   = row >> 11;
    const __bf16* q = qb + (size_t)row * CF_2D;
    const float*  ks = ksum + (size_t)bh * CF_2D;
    float s = 0.0f;
#pragma unroll
    for (int k = lane; k < CF_2D; k += 32) s += bf2f(q[k]) * ks[k];
#pragma unroll
    for (int off = 16; off; off >>= 1) s += __shfl_xor(s, off, 32);
    if (lane == 0) z[row] = 1.0f / fmaxf(s, 1e-3f);
}

// ---------------------------------------------------------------------------
// 5) attn[bh] = q_(Lx128) * kvt^T(128x64), scaled by z, stored bf16 (L,B,E).
//    One wave = 32x64 block (2x4 tiles); 128 heads x 64 strips = 8192 waves.
// ---------------------------------------------------------------------------
__global__ void attn_kernel(const __bf16* __restrict__ qb,
                            const __bf16* __restrict__ kvt,
                            const float* __restrict__ z,
                            __bf16* __restrict__ attnb) {
    const int lane = threadIdx.x & 31;
    const int wid  = blockIdx.x * 8 + (threadIdx.x >> 5);
    const int bh   = wid >> 6;     // 0..127
    const int ls   = wid & 63;     // 0..63 (32-row strip)

    const __bf16* A0  = qb  + (size_t)bh * CF_L * CF_2D
                            + (size_t)(ls * 32 +      (lane & 15)) * CF_2D + ((lane >> 4) * 8);
    const __bf16* A1  = A0 + (size_t)16 * CF_2D;
    const __bf16* Bt0 = kvt + (size_t)bh * CF_D * CF_2D
                            + (size_t)(lane & 15) * CF_2D + ((lane >> 4) * 8);

    v8f acc[2][4] = {{v8f{}, v8f{}, v8f{}, v8f{}}, {v8f{}, v8f{}, v8f{}, v8f{}}};
#pragma unroll
    for (int k0 = 0; k0 < CF_2D; k0 += 32) {
        v16bf a0 = load_a_bf(A0 + k0);
        v16bf a1 = load_a_bf(A1 + k0);
#pragma unroll
        for (int nt = 0; nt < 4; ++nt) {
            v16bf b = load_b_bf(Bt0 + (size_t)nt * 16 * CF_2D + k0);
            acc[0][nt] = WMMA_BF16(a0, b, acc[0][nt]);
            acc[1][nt] = WMMA_BF16(a1, b, acc[1][nt]);
        }
    }

    const int bb = bh >> 4, h = bh & 15;
#pragma unroll
    for (int mi = 0; mi < 2; ++mi) {
#pragma unroll
        for (int r = 0; r < 8; ++r) {
            const int l = ls * 32 + mi * 16 + ((lane < 16) ? r : 8 + r);
            const float zv = z[(size_t)bh * CF_L + l];
#pragma unroll
            for (int nt = 0; nt < 4; ++nt) {
                const int d = nt * 16 + (lane & 15);
                attnb[((size_t)l * CF_B + bb) * CF_E + h * CF_D + d] = f2bf(acc[mi][nt][r] * zv);
            }
        }
    }
}

// ---------------------------------------------------------------------------
// 6) out = attn(16384x1024) * out_w^T(1024x1024) + out_b  -> fp32
//    One wave = 32x64 block; 512 x 16 strips = 8192 waves.
// ---------------------------------------------------------------------------
__global__ void out_kernel(const __bf16* __restrict__ attnb,
                           const __bf16* __restrict__ W,
                           const float* __restrict__ bias,
                           float* __restrict__ out) {
    const int lane = threadIdx.x & 31;
    const int wid  = blockIdx.x * 8 + (threadIdx.x >> 5);
    const int ms   = wid >> 4;       // 0..511
    const int ns   = wid & 15;       // 0..15

    const __bf16* A0 = attnb + (size_t)(ms * 32 +      (lane & 15)) * CF_E + ((lane >> 4) * 8);
    const __bf16* A1 = attnb + (size_t)(ms * 32 + 16 + (lane & 15)) * CF_E + ((lane >> 4) * 8);
    const __bf16* Bt = W + (size_t)(ns * 64 + (lane & 15)) * CF_E + ((lane >> 4) * 8);

    v8f acc[2][4] = {{v8f{}, v8f{}, v8f{}, v8f{}}, {v8f{}, v8f{}, v8f{}, v8f{}}};
    for (int k0 = 0; k0 < CF_E; k0 += 32) {
        v16bf a0 = load_a_bf(A0 + k0);
        v16bf a1 = load_a_bf(A1 + k0);
#pragma unroll
        for (int nt = 0; nt < 4; ++nt) {
            v16bf b = load_b_bf(Bt + (size_t)nt * 16 * CF_E + k0);
            acc[0][nt] = WMMA_BF16(a0, b, acc[0][nt]);
            acc[1][nt] = WMMA_BF16(a1, b, acc[1][nt]);
        }
    }

#pragma unroll
    for (int nt = 0; nt < 4; ++nt) {
        const int col = ns * 64 + nt * 16 + (lane & 15);
        const float bv = bias[col];
#pragma unroll
        for (int mi = 0; mi < 2; ++mi) {
#pragma unroll
            for (int r = 0; r < 8; ++r) {
                const int row = ms * 32 + mi * 16 + ((lane < 16) ? r : 8 + r);
                out[(size_t)row * CF_E + col] = acc[mi][nt][r] + bv;
            }
        }
    }
}

// ---------------------------------------------------------------------------
extern "C" void kernel_launch(void* const* d_in, const int* in_sizes, int n_in,
                              void* d_out, int out_size, void* d_ws, size_t ws_size,
                              hipStream_t stream) {
    const float* query = (const float*)d_in[0];   // (L,B,E)
    const float* w_in  = (const float*)d_in[1];   // (3E,E)
    const float* b_in  = (const float*)d_in[2];   // (3E)
    const float* w_out = (const float*)d_in[3];   // (E,E)
    const float* b_out = (const float*)d_in[4];   // (E)
    float* out = (float*)d_out;

    char* ws = (char*)d_ws;
    size_t off = 0;
    __bf16* qb    = (__bf16*)(ws + off); off += (size_t)CF_BH * CF_L * CF_2D * 2;  // 64 MB
    __bf16* kt    = (__bf16*)(ws + off); off += (size_t)CF_BH * CF_2D * CF_L * 2;  // 64 MB
    __bf16* vt    = (__bf16*)(ws + off); off += (size_t)CF_BH * CF_D * CF_L * 2;   // 32 MB
    __bf16* attnb = (__bf16*)(ws + off); off += (size_t)CF_ROWS * CF_E * 2;        // 32 MB
    __bf16* xq    = (__bf16*)(ws + off); off += (size_t)CF_ROWS * CF_E * 2;        // 32 MB
    __bf16* wi_bf = (__bf16*)(ws + off); off += (size_t)CF_3E * CF_E * 2;          //  6 MB
    __bf16* wo_bf = (__bf16*)(ws + off); off += (size_t)CF_E * CF_E * 2;           //  2 MB
    __bf16* kvt   = (__bf16*)(ws + off); off += (size_t)CF_BH * CF_D * CF_2D * 2;  //  2 MB
    float*  ksum  = (float*)(ws + off);  off += (size_t)CF_BH * CF_2D * 4;
    float*  zbuf  = (float*)(ws + off);  off += (size_t)CF_BH * CF_L * 4;
    (void)ws_size; (void)in_sizes; (void)n_in; (void)out_size;

    // 0) one-time fp32 -> bf16 conversions (bandwidth bound)
    cvt_bf16_kernel<<<(CF_ROWS * CF_E) / (256 * 8), 256, 0, stream>>>(query, xq, CF_ROWS * CF_E);
    cvt_bf16_kernel<<<(CF_3E * CF_E) / (256 * 8), 256, 0, stream>>>(w_in, wi_bf, CF_3E * CF_E);
    cvt_bf16_kernel<<<(CF_E * CF_E) / (256 * 8), 256, 0, stream>>>(w_out, wo_bf, CF_E * CF_E);

    // 1) QKV projection + fused cosFormer reweighting (24576 waves)
    qkv_kernel<<<3072, 256, 0, stream>>>(xq, wi_bf, b_in, qb, kt, vt);
    // 2) k-sum reduction (16384 rows)
    ksum_kernel<<<2048, 256, 0, stream>>>(kt, ksum);
    // 3) per-head kv GEMM (128 heads, 8 waves each)
    kv_kernel<<<CF_BH, 256, 0, stream>>>(kt, vt, kvt);
    // 4) normalizer z (262144 rows)
    z_kernel<<<32768, 256, 0, stream>>>(qb, ksum, zbuf);
    // 5) per-head attn GEMM + z scaling (8192 waves)
    attn_kernel<<<1024, 256, 0, stream>>>(qb, kvt, zbuf, attnb);
    // 6) output projection (8192 waves)
    out_kernel<<<1024, 256, 0, stream>>>(attnb, wo_bf, b_out, out);
}
